// STConv_36146444763716
// MI455X (gfx1250) — matrix operand order repaired and checked by reading, not compiled, not run on hardware
//
#include <hip/hip_runtime.h>
#include <hip/hip_bf16.h>
#include <math.h>

#define NODES   304
#define TSTEPS  20
#define HID     64
#define GCN_IN  20
#define BN_EPS  1e-5f

typedef _Float16 half_t;
typedef _Float16 v16h __attribute__((ext_vector_type(16)));
typedef _Float16 v8h  __attribute__((ext_vector_type(8)));
typedef float    v8f  __attribute__((ext_vector_type(8)));

// ---------------------------------------------------------------------------
// GCN normalization helpers
// ---------------------------------------------------------------------------
__global__ void k_fill1(float* p, int n) {
    int i = blockIdx.x * blockDim.x + threadIdx.x;
    if (i < n) p[i] = 1.0f;   // self-loop weight
}

__global__ void k_deg_scatter(const int* __restrict__ col,
                              const float* __restrict__ ew,
                              float* __restrict__ deg, int E) {
    int i = blockIdx.x * blockDim.x + threadIdx.x;
    if (i < E) atomicAdd(&deg[col[i]], ew[i]);
}

__global__ void k_dinv(const float* __restrict__ deg, float* __restrict__ dinv, int n) {
    int i = blockIdx.x * blockDim.x + threadIdx.x;
    if (i < n) dinv[i] = rsqrtf(fmaxf(deg[i], 1e-12f));
}

__global__ void k_norm(const int* __restrict__ row, const int* __restrict__ col,
                       const float* __restrict__ ew, const float* __restrict__ dinv,
                       float* __restrict__ nrm, int E) {
    int i = blockIdx.x * blockDim.x + threadIdx.x;
    if (i < E) nrm[i] = dinv[row[i]] * ew[i] * dinv[col[i]];
}

// ---------------------------------------------------------------------------
// f16 packing kernels
// ---------------------------------------------------------------------------
__global__ void k_pack_a0(const float* __restrict__ x, half_t* __restrict__ A, int n) {
    int idx = blockIdx.x * blockDim.x + threadIdx.x;
    if (idx >= n * 32) return;
    int r = idx >> 5, k = idx & 31;
    A[idx] = (k < GCN_IN) ? (half_t)x[r * GCN_IN + k] : (half_t)0.0f;
}

// B[no][k] = (k<K && no<Nout) ? W[k][no] : 0 ; B is [Np, Kp] (transposed, padded)
__global__ void k_pack_wT(const float* __restrict__ W, half_t* __restrict__ B,
                          int K, int Nout, int Kp, int Np) {
    long long idx = (long long)blockIdx.x * blockDim.x + threadIdx.x;
    long long total = (long long)Np * Kp;
    if (idx >= total) return;
    int k  = (int)(idx / Np);
    int no = (int)(idx % Np);
    half_t v = (half_t)0.0f;
    if (k < K && no < Nout) v = (half_t)W[(long long)k * Nout + no];
    B[(long long)no * Kp + k] = v;
}

// ---------------------------------------------------------------------------
// Fragment loaders (contiguous-K lane layout, same for A and B)
// ---------------------------------------------------------------------------
// A (16x32 f16): lane holds row M=r; 16 contiguous K starting at k + hf*16
// -> one contiguous 32B load per lane.
static __device__ __forceinline__ v16h load_a_frag(const half_t* p) {
    return *(const v16h*)(p);
}

// ---------------------------------------------------------------------------
// Wave-level WMMA GEMM with NB-wide N-blocking and optional split-K.
// C[M,N] = A[M,Kp] * B^T[N,Kp]  (f16 in, f32 acc).
// Each wave owns a 16 x (16*NB) strip and one K-slice, software-pipelined
// one K-step deep with NB independent accumulators.
//   ksplit == 1 : full epilogue in-kernel (bias/relu, f32 and/or f16 store)
//   ksplit  > 1 : atomicAdd f32 partials into Cacc[M,Np]; epilogue separate
// Requirements: M % 16 == 0, Np % (16*NB) == 0, (Kp/ksplit) % 32 == 0.
// Wave index is forced scalar (readfirstlane) so tile ids / K bounds live in
// SGPRs -> scalar loop, and EXEC stays all-ones through the WMMA loop.
// ---------------------------------------------------------------------------
template <int NB>
__global__ void k_wmma_gemm(const half_t* __restrict__ A,
                            const half_t* __restrict__ B,
                            const float*  __restrict__ bias,
                            float*  __restrict__ Cf,
                            half_t* __restrict__ Ch,
                            float*  __restrict__ Cacc,
                            int M, int Np, int Kp, int Nout, int ldc,
                            int do_relu, int ksplit) {
    int lane = threadIdx.x & 31;
    int wv   = __builtin_amdgcn_readfirstlane(threadIdx.x >> 5);
    int wave = blockIdx.x * ((int)blockDim.x >> 5) + wv;
    int nblocks = Np / (16 * NB);
    int mtiles  = M >> 4;
    if (wave >= mtiles * nblocks * ksplit) return;     // uniform per wave
    int nb  = wave % nblocks;
    int tmp = wave / nblocks;
    int mt  = tmp % mtiles;
    int ks  = tmp / mtiles;
    int hf  = lane >> 4;      // lane-half: 0 => lanes 0-15, 1 => lanes 16-31
    int r   = lane & 15;

    int klen = Kp / ksplit;
    int k0   = ks * klen;

    const half_t* ap = A + (long long)(mt * 16 + r) * Kp + hf * 16;
    // B (32x16 f16): lane holds col N=r; 16 contiguous K starting at k + hf*16
    const half_t* bp = B + (long long)(nb * NB * 16 + r) * Kp + hf * 16;
    const long long btile = (long long)16 * Kp;        // halfs between N tiles

    v8f acc[NB];
#pragma unroll
    for (int j = 0; j < NB; ++j) acc[j] = (v8f){};

    // ---- prologue: fetch K-step k0 ----
    v16h a_c = load_a_frag(ap + k0);
    v16h b_c[NB];
#pragma unroll
    for (int j = 0; j < NB; ++j) b_c[j] = *(const v16h*)(bp + j * btile + k0);

    // ---- pipelined main loop: fetch k while multiplying k-32 ----
    for (int k = k0 + 32; k < k0 + klen; k += 32) {
        v16h a_n = load_a_frag(ap + k);
        v16h b_n[NB];
#pragma unroll
        for (int j = 0; j < NB; ++j) b_n[j] = *(const v16h*)(bp + j * btile + k);
#pragma unroll
        for (int j = 0; j < NB; ++j)
            acc[j] = __builtin_amdgcn_wmma_f32_16x16x32_f16(false, a_c, false, b_c[j],
                                                            (short)0, acc[j], false, false);
        a_c = a_n;
#pragma unroll
        for (int j = 0; j < NB; ++j) b_c[j] = b_n[j];
    }
    // ---- drain ----
#pragma unroll
    for (int j = 0; j < NB; ++j)
        acc[j] = __builtin_amdgcn_wmma_f32_16x16x32_f16(false, a_c, false, b_c[j],
                                                        (short)0, acc[j], false, false);

    // ---- epilogue ----
#pragma unroll
    for (int j = 0; j < NB; ++j) {
        int col = nb * NB * 16 + j * 16 + r;
        if (ksplit > 1) {
#pragma unroll
            for (int i = 0; i < 8; ++i) {
                int rowo = mt * 16 + i + hf * 8;   // C layout: VGPR i -> M=i (+8 hi half)
                atomicAdd(&Cacc[(long long)rowo * Np + col], acc[j][i]);
            }
        } else if (col < Nout) {
            float bv = bias ? bias[col] : 0.0f;
#pragma unroll
            for (int i = 0; i < 8; ++i) {
                int rowo = mt * 16 + i + hf * 8;
                float v = acc[j][i] + bv;
                if (do_relu) v = fmaxf(v, 0.0f);
                if (Cf) Cf[(long long)rowo * ldc + col] = v;
                if (Ch) Ch[(long long)rowo * ldc + col] = (half_t)v;
            }
        }
    }
}

// split-K epilogue: bias + optional relu + f32/f16 store from Cacc[M,Np]
__global__ void k_epilogue(const float* __restrict__ Cacc, const float* __restrict__ bias,
                           float* __restrict__ Cf, half_t* __restrict__ Ch,
                           int M, int Np, int Nout, int ldc, int do_relu) {
    long long idx = (long long)blockIdx.x * blockDim.x + threadIdx.x;
    if (idx >= (long long)M * Np) return;
    int rowo = (int)(idx / Np);
    int col  = (int)(idx % Np);
    if (col >= Nout) return;
    float v = Cacc[idx] + (bias ? bias[col] : 0.0f);
    if (do_relu) v = fmaxf(v, 0.0f);
    if (Cf) Cf[(long long)rowo * ldc + col] = v;
    if (Ch) Ch[(long long)rowo * ldc + col] = (half_t)v;
}

// ---------------------------------------------------------------------------
// Edge scatter: agg[col][c] += h[row][c] * norm[e]
// ---------------------------------------------------------------------------
__global__ void k_scatter(const int* __restrict__ row, const int* __restrict__ col,
                          const float* __restrict__ nrm, const float* __restrict__ h,
                          float* __restrict__ agg, int E) {
    long long idx = (long long)blockIdx.x * blockDim.x + threadIdx.x;
    if (idx >= (long long)E * HID) return;
    int e = (int)(idx >> 6);
    int c = (int)(idx & (HID - 1));
    atomicAdd(&agg[(long long)col[e] * HID + c], h[(long long)row[e] * HID + c] * nrm[e]);
}

// post-GCN: x = relu(bn(agg + h*dinv^2 + bias))  -> f16
__global__ void k_post_gcn(const float* __restrict__ agg, const float* __restrict__ h,
                           const float* __restrict__ dinv, const float* __restrict__ bias,
                           const float* __restrict__ g, const float* __restrict__ bb,
                           half_t* __restrict__ out, int n) {
    long long idx = (long long)blockIdx.x * blockDim.x + threadIdx.x;
    if (idx >= (long long)n * HID) return;
    int i = (int)(idx >> 6);
    int c = (int)(idx & (HID - 1));
    float d = dinv[i];
    float v = agg[idx] + h[idx] * d * d + bias[c];
    v = v * (g[c] * rsqrtf(1.0f + BN_EPS)) + bb[c];
    out[idx] = (half_t)fmaxf(v, 0.0f);
}

// ---------------------------------------------------------------------------
// Temporal conv 1 (1 -> 64 ch, gated GLU, k=3 over T, pad 1), ReLU,
// writing directly into Tcat slots t = 0..19 (f16).
// ---------------------------------------------------------------------------
__global__ void k_tc1(const float* __restrict__ x,
                      const float* __restrict__ w1, const float* __restrict__ b1,
                      const float* __restrict__ w2, const float* __restrict__ b2,
                      const float* __restrict__ w3, const float* __restrict__ b3,
                      half_t* __restrict__ Tcat, int b) {
    long long idx = (long long)blockIdx.x * blockDim.x + threadIdx.x;
    long long total = (long long)b * TSTEPS * NODES * HID;
    if (idx >= total) return;
    int co   = (int)(idx & (HID - 1));
    long long q = idx >> 6;
    int node = (int)(q % NODES);  q /= NODES;
    int t    = (int)(q % TSTEPS);
    int bi   = (int)(q / TSTEPS);

    long long base = (long long)bi * (TSTEPS * NODES) + node;
    float xm = (t > 0)          ? x[base + (long long)(t - 1) * NODES] : 0.0f;
    float x0 =                    x[base + (long long)t * NODES];
    float xp = (t < TSTEPS - 1) ? x[base + (long long)(t + 1) * NODES] : 0.0f;

    float P = b1[co] + w1[co * 3] * xm + w1[co * 3 + 1] * x0 + w1[co * 3 + 2] * xp;
    float Q = b2[co] + w2[co * 3] * xm + w2[co * 3 + 1] * x0 + w2[co * 3 + 2] * xp;
    float R = b3[co] + w3[co * 3] * xm + w3[co * 3 + 1] * x0 + w3[co * 3 + 2] * xp;
    float H = fmaxf(P * (1.0f / (1.0f + __expf(-Q))) + R, 0.0f);

    Tcat[(((long long)bi * 21 + t) * NODES + node) * HID + co] = (half_t)H;
}

// gfeat = x2.T.reshape(b,1,NODES,HID) into Tcat slot t=20 (x2 >= 0, relu no-op)
__global__ void k_gfeat(const half_t* __restrict__ x2, half_t* __restrict__ Tcat,
                        int b, int n) {
    long long idx = (long long)blockIdx.x * blockDim.x + threadIdx.x;
    long long total = (long long)b * NODES * HID;
    if (idx >= total) return;
    int h2   = (int)(idx & (HID - 1));
    long long q = idx >> 6;
    int node = (int)(q % NODES);
    int bi   = (int)(q / NODES);
    long long f = (long long)bi * (NODES * HID) + node * HID + h2;  // flat idx of x.T
    int hh = (int)(f / n);
    int i  = (int)(f % n);
    Tcat[(((long long)bi * 21 + 20) * NODES + node) * HID + h2] = x2[(long long)i * HID + hh];
}

// ---------------------------------------------------------------------------
// Temporal conv 2 (64 -> 1 ch, gated, k=3 over the 21-slot axis) + BN2d(node)
// -> Tflat f16 [b, 6400] (cols 6384..6399 pre-zeroed by memset)
// ---------------------------------------------------------------------------
__global__ void k_tc2(const half_t* __restrict__ Tcat,
                      const float* __restrict__ w1, const float* __restrict__ b1,
                      const float* __restrict__ w2, const float* __restrict__ b2,
                      const float* __restrict__ w3, const float* __restrict__ b3,
                      const float* __restrict__ bng, const float* __restrict__ bnb,
                      half_t* __restrict__ Tflat, int b) {
    __shared__ float sw[3 * HID * 3];   // [conv][c*3+kw]
    for (int j = threadIdx.x; j < HID * 3; j += blockDim.x) {
        sw[j]               = w1[j];
        sw[HID * 3 + j]     = w2[j];
        sw[2 * HID * 3 + j] = w3[j];
    }
    __syncthreads();

    long long idx = (long long)blockIdx.x * blockDim.x + threadIdx.x;
    long long total = (long long)b * 21 * NODES;
    if (idx >= total) return;
    int node = (int)(idx % NODES);
    long long q = idx / NODES;
    int t  = (int)(q % 21);
    int bi = (int)(q / 21);

    float a1 = 0.0f, a2 = 0.0f, a3 = 0.0f;
#pragma unroll
    for (int dt = 0; dt < 3; ++dt) {
        int tt = t + dt - 1;
        if (tt < 0 || tt >= 21) continue;
        const half_t* p = Tcat + (((long long)bi * 21 + tt) * NODES + node) * HID;
#pragma unroll 8
        for (int c = 0; c < HID; ++c) {
            float v = (float)p[c];
            a1 += v * sw[c * 3 + dt];
            a2 += v * sw[HID * 3 + c * 3 + dt];
            a3 += v * sw[2 * HID * 3 + c * 3 + dt];
        }
    }
    float P = a1 + b1[0], Q = a2 + b2[0], R = a3 + b3[0];
    float H = fmaxf(P * (1.0f / (1.0f + __expf(-Q))) + R, 0.0f);
    float y = H * (bng[node] * rsqrtf(1.0f + BN_EPS)) + bnb[node];
    Tflat[(long long)bi * 6400 + t * NODES + node] = (half_t)y;
}

// ---------------------------------------------------------------------------
// Host-side launcher
// ---------------------------------------------------------------------------
static inline unsigned cdiv_ll(long long a, long long b) { return (unsigned)((a + b - 1) / b); }

extern "C" void kernel_launch(void* const* d_in, const int* in_sizes, int n_in,
                              void* d_out, int out_size, void* d_ws, size_t ws_size,
                              hipStream_t stream) {
    const float* x      = (const float*)d_in[0];
    const int*   ei     = (const int*)d_in[1];
    const float* ew     = (const float*)d_in[2];
    const float* tc1w1  = (const float*)d_in[3];
    const float* tc1b1  = (const float*)d_in[4];
    const float* tc1w2  = (const float*)d_in[5];
    const float* tc1b2  = (const float*)d_in[6];
    const float* tc1w3  = (const float*)d_in[7];
    const float* tc1b3  = (const float*)d_in[8];
    const float* gW0    = (const float*)d_in[9];
    const float* gb0    = (const float*)d_in[10];
    const float* bn0g   = (const float*)d_in[11];
    const float* bn0b   = (const float*)d_in[12];
    const float* gW1    = (const float*)d_in[13];
    const float* gb1    = (const float*)d_in[14];
    const float* bn1g   = (const float*)d_in[15];
    const float* bn1b   = (const float*)d_in[16];
    const float* tc2w1  = (const float*)d_in[17];
    const float* tc2b1  = (const float*)d_in[18];
    const float* tc2w2  = (const float*)d_in[19];
    const float* tc2b2  = (const float*)d_in[20];
    const float* tc2w3  = (const float*)d_in[21];
    const float* tc2b3  = (const float*)d_in[22];
    const float* bn2g   = (const float*)d_in[23];
    const float* bn2b   = (const float*)d_in[24];
    const float* fcW    = (const float*)d_in[25];
    const float* fcb    = (const float*)d_in[26];
    const float* fc4W   = (const float*)d_in[27];
    const float* fc4b   = (const float*)d_in[28];

    const int n = in_sizes[0] / GCN_IN;       // 38912
    const int E = in_sizes[1] / 2;            // 622592
    const int b = n / NODES;                  // 128
    const int* row = ei;
    const int* col = ei + E;

    // bump allocator on d_ws
    char*  ws  = (char*)d_ws;
    size_t off = 0;
    auto alloc = [&](size_t bytes) -> void* {
        void* p = ws + off;
        off = (off + bytes + 255) & ~(size_t)255;
        return p;
    };
    float*  deg   = (float*)alloc((size_t)n * 4);
    float*  dinv  = (float*)alloc((size_t)n * 4);
    float*  nrm   = (float*)alloc((size_t)E * 4);
    half_t* A0    = (half_t*)alloc((size_t)n * 32 * 2);
    half_t* B0    = (half_t*)alloc((size_t)64 * 32 * 2);
    float*  hbuf  = (float*)alloc((size_t)n * HID * 4);
    float*  agg   = (float*)alloc((size_t)n * HID * 4);
    half_t* X1h   = (half_t*)alloc((size_t)n * HID * 2);
    half_t* B1    = (half_t*)alloc((size_t)64 * 64 * 2);
    half_t* X2h   = (half_t*)alloc((size_t)n * HID * 2);
    half_t* Tcat  = (half_t*)alloc((size_t)b * 21 * NODES * HID * 2);
    half_t* Tflat = (half_t*)alloc((size_t)b * 6400 * 2);
    half_t* BWfc  = (half_t*)alloc((size_t)2048 * 6400 * 2);
    half_t* Afc   = (half_t*)alloc((size_t)b * 2048 * 2);
    half_t* BW4   = (half_t*)alloc((size_t)304 * 2048 * 2);
    float*  Cacc  = (float*)alloc((size_t)b * 2048 * 4);   // split-K accumulator

    const int LB = 256;

    // ---- GCN normalization ----
    k_fill1<<<cdiv_ll(n, LB), LB, 0, stream>>>(deg, n);
    k_deg_scatter<<<cdiv_ll(E, LB), LB, 0, stream>>>(col, ew, deg, E);
    k_dinv<<<cdiv_ll(n, LB), LB, 0, stream>>>(deg, dinv, n);
    k_norm<<<cdiv_ll(E, LB), LB, 0, stream>>>(row, col, ew, dinv, nrm, E);

    // ---- GCN layer 0 : h = x @ W0  (M=n, N=64, K=32; NB=4 covers N in 1 wave)
    k_pack_a0<<<cdiv_ll((long long)n * 32, LB), LB, 0, stream>>>(x, A0, n);
    k_pack_wT<<<cdiv_ll((long long)64 * 32, LB), LB, 0, stream>>>(gW0, B0, GCN_IN, HID, 32, 64);
    {
        long long waves = (long long)(n >> 4) * 1;
        k_wmma_gemm<4><<<cdiv_ll(waves * 32, LB), LB, 0, stream>>>(
            A0, B0, nullptr, hbuf, nullptr, nullptr, n, 64, 32, 64, 64, 0, 1);
    }
    hipMemsetAsync(agg, 0, (size_t)n * HID * 4, stream);
    k_scatter<<<cdiv_ll((long long)E * HID, LB), LB, 0, stream>>>(row, col, nrm, hbuf, agg, E);
    k_post_gcn<<<cdiv_ll((long long)n * HID, LB), LB, 0, stream>>>(agg, hbuf, dinv, gb0, bn0g, bn0b, X1h, n);

    // ---- GCN layer 1 : h = x1 @ W1  (M=n, N=64, K=64)
    k_pack_wT<<<cdiv_ll((long long)64 * 64, LB), LB, 0, stream>>>(gW1, B1, HID, HID, 64, 64);
    {
        long long waves = (long long)(n >> 4) * 1;
        k_wmma_gemm<4><<<cdiv_ll(waves * 32, LB), LB, 0, stream>>>(
            X1h, B1, nullptr, hbuf, nullptr, nullptr, n, 64, 64, 64, 64, 0, 1);
    }
    hipMemsetAsync(agg, 0, (size_t)n * HID * 4, stream);
    k_scatter<<<cdiv_ll((long long)E * HID, LB), LB, 0, stream>>>(row, col, nrm, hbuf, agg, E);
    k_post_gcn<<<cdiv_ll((long long)n * HID, LB), LB, 0, stream>>>(agg, hbuf, dinv, gb1, bn1g, bn1b, X2h, n);

    // ---- temporal conv 1 -> Tcat[t=0..19]; gfeat -> Tcat[t=20] ----
    k_tc1<<<cdiv_ll((long long)b * TSTEPS * NODES * HID, LB), LB, 0, stream>>>(
        x, tc1w1, tc1b1, tc1w2, tc1b2, tc1w3, tc1b3, Tcat, b);
    k_gfeat<<<cdiv_ll((long long)b * NODES * HID, LB), LB, 0, stream>>>(X2h, Tcat, b, n);

    // ---- temporal conv 2 + BN2d -> Tflat [b, 6400] ----
    hipMemsetAsync(Tflat, 0, (size_t)b * 6400 * 2, stream);
    k_tc2<<<cdiv_ll((long long)b * 21 * NODES, LB), LB, 0, stream>>>(
        Tcat, tc2w1, tc2b1, tc2w2, tc2b2, tc2w3, tc2b3, bn2g, bn2b, Tflat, b);

    // ---- FC1: [b,6400] @ [6400,2048] + bias, relu -> f16 (split-K=8, NB=4) ----
    k_pack_wT<<<cdiv_ll((long long)2048 * 6400, LB), LB, 0, stream>>>(fcW, BWfc, 6384, 2048, 6400, 2048);
    hipMemsetAsync(Cacc, 0, (size_t)b * 2048 * 4, stream);
    {
        long long waves = (long long)(b >> 4) * (2048 / 64) * 8;
        k_wmma_gemm<4><<<cdiv_ll(waves * 32, LB), LB, 0, stream>>>(
            Tflat, BWfc, nullptr, nullptr, nullptr, Cacc, b, 2048, 6400, 2048, 2048, 1, 8);
    }
    k_epilogue<<<cdiv_ll((long long)b * 2048, LB), LB, 0, stream>>>(
        Cacc, fcb, nullptr, Afc, b, 2048, 2048, 2048, 1);

    // ---- FC4: [b,2048] @ [2048,300] + bias -> d_out f32 (split-K=4, NB=1) ----
    k_pack_wT<<<cdiv_ll((long long)304 * 2048, LB), LB, 0, stream>>>(fc4W, BW4, 2048, 300, 2048, 304);
    hipMemsetAsync(Cacc, 0, (size_t)b * 304 * 4, stream);
    {
        long long waves = (long long)(b >> 4) * (304 / 16) * 4;
        k_wmma_gemm<1><<<cdiv_ll(waves * 32, LB), LB, 0, stream>>>(
            Afc, BW4, nullptr, nullptr, nullptr, Cacc, b, 304, 2048, 300, 300, 0, 4);
    }
    k_epilogue<<<cdiv_ll((long long)b * 304, LB), LB, 0, stream>>>(
        Cacc, fc4b, (float*)d_out, nullptr, b, 304, 300, 300, 0);
}
